// LlamaAttention_heavy_hitter_65970697667353
// MI455X (gfx1250) — matrix operand order repaired and checked by reading, not compiled, not run on hardware
//
#include <hip/hip_runtime.h>
#include <hip/hip_bf16.h>
#include <math.h>

#define S_LEN 2048
#define HID   4096
#define NH    32
#define NKV   8
#define HD    128
#define QKV_N 6144      // 4096 (Q) + 1024 (K) + 1024 (V)
#define KV_DIM 1024
#define RECENT 204
#define HEAVY  819
#define MASK_LEN 2049   // S + 1
#define TOPK_RANGE 1844 // S - RECENT
#define SCALE_F 0.08838834764831845f // 1/sqrt(128)
#define PSTR 40         // LDS P-tile row stride (elements), 16B-aligned, conflict-padded

typedef __bf16 bf16_t;
typedef __attribute__((ext_vector_type(8)))  __bf16 v8bf;
typedef __attribute__((ext_vector_type(16))) __bf16 v16bf;
typedef __attribute__((ext_vector_type(8)))  float  v8f;

static __device__ __forceinline__ v8f wmma_bf16(v16bf a, v16bf b, v8f c) {
  return __builtin_amdgcn_wmma_f32_16x16x32_bf16(false, a, false, b, (short)0, c, false, false);
}

// A-matrix 16x32 bf16 fragment: lanes 0-15 row M=lane, elems K=0..15;
// lanes 16-31 row M=lane-16, elems K=16..31.  rowptr = &A[M][k_chunk_base].
static __device__ __forceinline__ v16bf load_a16(const bf16_t* rowptr, int lane) {
  const int off = (lane & 16) ? 16 : 0;
  v8bf lo = *(const v8bf*)(rowptr + off);
  v8bf hi = *(const v8bf*)(rowptr + off + 8);
  v16bf a;
#pragma unroll
  for (int i = 0; i < 8; ++i) { a[i] = lo[i]; a[i + 8] = hi[i]; }
  return a;
}

// B-matrix 32x16 bf16 fragment: lanes 0-15 col N=lane, K=0..15; lanes 16-31 col N=lane-16,
// K=16..31.  rowptr = &Bt[N][k_chunk_base] with Bt stored N-major (so K is contiguous).
static __device__ __forceinline__ v16bf load_b16(const bf16_t* rowptr, int lane) {
  return *(const v16bf*)(rowptr + ((lane & 16) ? 16 : 0));
}

// ---------------- conversion kernels ----------------
__global__ void cvt_bf16_kernel(const float* __restrict__ src, bf16_t* __restrict__ dst, int n) {
  int i = blockIdx.x * blockDim.x + threadIdx.x;
  if (i < n) dst[i] = (bf16_t)src[i];
}

// src: K x N fp32 row-major  ->  dst: N x K bf16 row-major (transposed)
__global__ void cvt_transpose_kernel(const float* __restrict__ src, bf16_t* __restrict__ dst,
                                     int K, int N) {
  int i = blockIdx.x * blockDim.x + threadIdx.x;
  if (i >= K * N) return;
  int k = i / N, n = i % N;
  dst[(size_t)n * K + k] = (bf16_t)src[i];
}

// V slice of QKV (cols 5120..6143) -> Vt bf16 (1024 x 2048), d-major for PV B-fragments
__global__ void cvt_vt_kernel(const float* __restrict__ qkv, bf16_t* __restrict__ vt) {
  int i = blockIdx.x * blockDim.x + threadIdx.x;
  if (i >= S_LEN * KV_DIM) return;
  int q = i / KV_DIM, c = i % KV_DIM;
  vt[(size_t)c * S_LEN + q] = (bf16_t)qkv[(size_t)q * QKV_N + 5120 + c];
}

// ---------------- RoPE ----------------
__global__ void rope_kernel(const float* __restrict__ qkv, bf16_t* __restrict__ qbf,
                            bf16_t* __restrict__ kbf) {
  int i = blockIdx.x * blockDim.x + threadIdx.x;
  const int total = S_LEN * (NH + NKV) * (HD / 2);
  if (i >= total) return;
  int d = i % (HD / 2);
  int t = i / (HD / 2);
  int head = t % (NH + NKV);
  int pos  = t / (NH + NKV);
  float inv = __powf(10000.0f, -(float)(2 * d) * (1.0f / 128.0f));
  float fr = (float)pos * inv;
  float s, c;
  __sincosf(fr, &s, &c);
  if (head < NH) {
    const float* base = qkv + (size_t)pos * QKV_N + head * HD;
    float x0 = base[d], x1 = base[d + HD / 2];
    bf16_t* ob = qbf + (size_t)pos * HID + head * HD;
    ob[d]          = (bf16_t)(x0 * c - x1 * s);
    ob[d + HD / 2] = (bf16_t)(x1 * c + x0 * s);
  } else {
    int kh = head - NH;
    const float* base = qkv + (size_t)pos * QKV_N + HID + kh * HD;
    float x0 = base[d], x1 = base[d + HD / 2];
    bf16_t* ob = kbf + (size_t)pos * KV_DIM + kh * HD;
    ob[d]          = (bf16_t)(x0 * c - x1 * s);
    ob[d + HD / 2] = (bf16_t)(x1 * c + x0 * s);
  }
}

// ---------------- bf16 WMMA GEMM: C(MxN) = A(MxK) * Bt(NxK)^T ----------------
// block = 8 waves, 128(M) x 64(N) tile; each wave: 16 rows x 4 16-wide column tiles.
__global__ void __launch_bounds__(256) gemm_bf16_kernel(const bf16_t* __restrict__ A,
                                                        const bf16_t* __restrict__ Bt,
                                                        float* __restrict__ C,
                                                        int M, int N, int K) {
  const int lane  = threadIdx.x & 31;
  const int wave  = threadIdx.x >> 5;
  const int rlane = lane & 15;
  const int row0  = blockIdx.x * 128 + wave * 16;
  const int col0  = blockIdx.y * 64;
  v8f acc[4];
#pragma unroll
  for (int t = 0; t < 4; ++t)
#pragma unroll
    for (int e = 0; e < 8; ++e) acc[t][e] = 0.0f;
  const bf16_t* arow = A + (size_t)(row0 + rlane) * K;
  for (int k0 = 0; k0 < K; k0 += 32) {
    v16bf a = load_a16(arow + k0, lane);
#pragma unroll
    for (int t = 0; t < 4; ++t) {
      const bf16_t* brow = Bt + (size_t)(col0 + t * 16 + rlane) * K + k0;
      acc[t] = wmma_bf16(a, load_b16(brow, lane), acc[t]);
    }
  }
  const int rbase = row0 + ((lane & 16) ? 8 : 0);
#pragma unroll
  for (int t = 0; t < 4; ++t)
#pragma unroll
    for (int e = 0; e < 8; ++e)
      C[(size_t)(rbase + e) * N + (col0 + t * 16 + rlane)] = acc[t][e];
}

// ---------------- attention pass 1: per-row max + softmax denominator ----------------
__global__ void __launch_bounds__(256) attn_pass1_kernel(const bf16_t* __restrict__ Qbf,
                                                         const bf16_t* __restrict__ Kbf,
                                                         float* __restrict__ m_buf,
                                                         float* __restrict__ l_buf) {
  const int lane  = threadIdx.x & 31;
  const int wave  = threadIdx.x >> 5;
  const int rlane = lane & 15;
  const int h     = blockIdx.x >> 4;   // 16 query blocks of 128
  const int qblk  = blockIdx.x & 15;
  const int kvh   = h >> 2;
  const int qbase = qblk * 128 + wave * 16;
  const int rowoff = (lane & 16) ? 8 : 0;

  v16bf aq[4];
  const bf16_t* qrow = Qbf + (size_t)(qbase + rlane) * HID + h * HD;
#pragma unroll
  for (int f = 0; f < 4; ++f) aq[f] = load_a16(qrow + f * 32, lane);

  float mrun[8], lrun[8];
#pragma unroll
  for (int e = 0; e < 8; ++e) { mrun[e] = -3.0e38f; lrun[e] = 0.0f; }

  const int kmax = qbase + 16;
  for (int kb = 0; kb < kmax; kb += 16) {
    v8f c;
#pragma unroll
    for (int e = 0; e < 8; ++e) c[e] = 0.0f;
    const bf16_t* krow = Kbf + (size_t)(kb + rlane) * KV_DIM + kvh * HD;
#pragma unroll
    for (int f = 0; f < 4; ++f) c = wmma_bf16(aq[f], load_b16(krow + f * 32, lane), c);
    const int kidx = kb + rlane;
#pragma unroll
    for (int e = 0; e < 8; ++e) {
      const int qidx = qbase + rowoff + e;
      const bool valid = (kidx <= qidx);
      float s = valid ? c[e] * SCALE_F : -3.0e38f;
      float rmax = s;
#pragma unroll
      for (int off = 1; off < 16; off <<= 1) rmax = fmaxf(rmax, __shfl_xor(rmax, off, 32));
      float newm = fmaxf(mrun[e], rmax);
      float p = valid ? __expf(s - newm) : 0.0f;
#pragma unroll
      for (int off = 1; off < 16; off <<= 1) p += __shfl_xor(p, off, 32);
      lrun[e] = lrun[e] * __expf(mrun[e] - newm) + p;
      mrun[e] = newm;
    }
  }
  if (rlane == 0) {
#pragma unroll
    for (int e = 0; e < 8; ++e) {
      m_buf[h * S_LEN + qbase + rowoff + e] = mrun[e];
      l_buf[h * S_LEN + qbase + rowoff + e] = lrun[e];
    }
  }
}

// ---------------- attention pass 2: normalized P, P*V, per-key column sums ----------------
__global__ void __launch_bounds__(256) attn_pass2_kernel(const bf16_t* __restrict__ Qbf,
                                                         const bf16_t* __restrict__ Kbf,
                                                         const bf16_t* __restrict__ Vt,
                                                         const float* __restrict__ m_buf,
                                                         const float* __restrict__ l_buf,
                                                         bf16_t* __restrict__ Obf,
                                                         float* __restrict__ colsum_part) {
  __shared__ bf16_t pl[8 * 16 * PSTR];
  const int lane  = threadIdx.x & 31;
  const int wave  = threadIdx.x >> 5;
  const int rlane = lane & 15;
  const int h     = blockIdx.x >> 4;
  const int qblk  = blockIdx.x & 15;
  const int kvh   = h >> 2;
  const int qbase = qblk * 128 + wave * 16;
  const int rowoff = (lane & 16) ? 8 : 0;
  bf16_t* pw = pl + wave * 16 * PSTR;

  v16bf aq[4];
  const bf16_t* qrow = Qbf + (size_t)(qbase + rlane) * HID + h * HD;
#pragma unroll
  for (int f = 0; f < 4; ++f) aq[f] = load_a16(qrow + f * 32, lane);

  float mrow[8], linv[8];
#pragma unroll
  for (int e = 0; e < 8; ++e) {
    mrow[e] = m_buf[h * S_LEN + qbase + rowoff + e];
    linv[e] = 1.0f / l_buf[h * S_LEN + qbase + rowoff + e];
  }

  v8f o[8];
#pragma unroll
  for (int t = 0; t < 8; ++t)
#pragma unroll
    for (int e = 0; e < 8; ++e) o[t][e] = 0.0f;

  float* csrow = colsum_part + (size_t)(h * 128 + qblk * 8 + wave) * S_LEN;
  const int kmax = qbase + 16;
  for (int kb = 0; kb < kmax; kb += 32) {
#pragma unroll
    for (int sub = 0; sub < 2; ++sub) {
      const int ks = kb + sub * 16;
      float colsum = 0.0f;
      if (ks < kmax) {
        v8f c;
#pragma unroll
        for (int e = 0; e < 8; ++e) c[e] = 0.0f;
        const bf16_t* krow = Kbf + (size_t)(ks + rlane) * KV_DIM + kvh * HD;
#pragma unroll
        for (int f = 0; f < 4; ++f) c = wmma_bf16(aq[f], load_b16(krow + f * 32, lane), c);
        const int kidx = ks + rlane;
#pragma unroll
        for (int e = 0; e < 8; ++e) {
          const int qidx = qbase + rowoff + e;
          float p = (kidx <= qidx) ? __expf(c[e] * SCALE_F - mrow[e]) * linv[e] : 0.0f;
          colsum += p;
          pw[(rowoff + e) * PSTR + sub * 16 + rlane] = (bf16_t)p;
        }
        colsum += __shfl_xor(colsum, 16, 32);
        if (lane < 16) csrow[ks + lane] = colsum;   // deterministic per-wave partial
      } else {
#pragma unroll
        for (int e = 0; e < 8; ++e)
          pw[(rowoff + e) * PSTR + sub * 16 + rlane] = (bf16_t)0.0f;
      }
    }
    __builtin_amdgcn_wave_barrier();
    // re-load P (16x32, C-layout in LDS) as an A-fragment for P*V
    v16bf pa = load_a16(pw + rlane * PSTR, lane);
#pragma unroll
    for (int t = 0; t < 8; ++t) {
      const bf16_t* vrow = Vt + (size_t)(kvh * HD + t * 16 + rlane) * S_LEN + kb;
      o[t] = wmma_bf16(pa, load_b16(vrow, lane), o[t]);
    }
    __builtin_amdgcn_wave_barrier();
  }
  // zero never-visited key slots in this wave's partial-sum row
  if (lane < 16)
    for (int k = kmax + lane; k < S_LEN; k += 16) csrow[k] = 0.0f;
  // write O tile as bf16 for the Wo GEMM
#pragma unroll
  for (int t = 0; t < 8; ++t)
#pragma unroll
    for (int e = 0; e < 8; ++e)
      Obf[(size_t)(qbase + rowoff + e) * HID + h * HD + t * 16 + rlane] = (bf16_t)o[t][e];
}

// ---------------- reduce 128 per-wave partials -> current_scores[h][k] ----------------
__global__ void reduce_colsum_kernel(const float* __restrict__ part, float* __restrict__ cur) {
  int i = blockIdx.x * blockDim.x + threadIdx.x;
  if (i >= NH * S_LEN) return;
  int h = i / S_LEN, k = i % S_LEN;
  const float* p = part + (size_t)h * 128 * S_LEN + k;
  float s = 0.0f;
  for (int j = 0; j < 128; ++j) s += p[(size_t)j * S_LEN];
  cur[i] = s;
}

// ---------------- heavy-hitter mask: top-HEAVY of first 1844 keys + last RECENT ----------------
__global__ void __launch_bounds__(256) topk_mask_kernel(const float* __restrict__ cur,
                                                        float* __restrict__ mask_out) {
  __shared__ float sv[TOPK_RANGE];
  __shared__ float rmax[256];
  __shared__ int   ridx[256];
  const int h = blockIdx.x;
  const int tid = threadIdx.x;
  for (int i = tid; i < TOPK_RANGE; i += 256) sv[i] = cur[h * S_LEN + i];
  float* mrow = mask_out + (size_t)h * MASK_LEN;
  for (int j = tid; j < MASK_LEN; j += 256)
    mrow[j] = (j >= MASK_LEN - RECENT) ? 1.0f : 0.0f;
  __syncthreads();
  for (int it = 0; it < HEAVY; ++it) {
    float best = -3.0e38f; int bi = TOPK_RANGE;
    for (int i = tid; i < TOPK_RANGE; i += 256) {
      float v = sv[i];
      if (v > best) { best = v; bi = i; }
    }
    rmax[tid] = best; ridx[tid] = bi;
    __syncthreads();
    for (int s = 128; s > 0; s >>= 1) {
      if (tid < s) {
        if (rmax[tid + s] > rmax[tid] ||
            (rmax[tid + s] == rmax[tid] && ridx[tid + s] < ridx[tid])) {
          rmax[tid] = rmax[tid + s]; ridx[tid] = ridx[tid + s];
        }
      }
      __syncthreads();
    }
    if (tid == 0) { mrow[ridx[0]] = 1.0f; sv[ridx[0]] = -3.4e38f; }
    __syncthreads();
  }
}

extern "C" void kernel_launch(void* const* d_in, const int* in_sizes, int n_in,
                              void* d_out, int out_size, void* d_ws, size_t ws_size,
                              hipStream_t stream) {
  const float* X  = (const float*)d_in[0];
  // d_in[1] = position_ids (arange) — positions derived from indices directly
  const float* Wq = (const float*)d_in[2];
  const float* Wk = (const float*)d_in[3];
  const float* Wv = (const float*)d_in[4];
  const float* Wo = (const float*)d_in[5];
  float* out = (float*)d_out;

  char* ws = (char*)d_ws;
  size_t o = 0;
  bf16_t* Xbf    = (bf16_t*)(ws + o); o += (size_t)S_LEN * HID * 2;
  bf16_t* Wqkv_t = (bf16_t*)(ws + o); o += (size_t)QKV_N * HID * 2;  // fused Wq|Wk|Wv, N-major
  bf16_t* Wo_t   = (bf16_t*)(ws + o); o += (size_t)HID * HID * 2;
  float*  QKV    = (float*) (ws + o); o += (size_t)S_LEN * QKV_N * 4;
  bf16_t* Qbf    = (bf16_t*)(ws + o); o += (size_t)S_LEN * HID * 2;
  bf16_t* Kbf    = (bf16_t*)(ws + o); o += (size_t)S_LEN * KV_DIM * 2;
  bf16_t* Vt     = (bf16_t*)(ws + o); o += (size_t)KV_DIM * S_LEN * 2;
  float*  m_buf  = (float*) (ws + o); o += (size_t)NH * S_LEN * 4;
  float*  l_buf  = (float*) (ws + o); o += (size_t)NH * S_LEN * 4;
  float*  cspart = (float*) (ws + o); o += (size_t)NH * 128 * S_LEN * 4;
  float*  cur    = (float*) (ws + o); o += (size_t)NH * S_LEN * 4;
  bf16_t* Obf    = (bf16_t*)(ws + o); o += (size_t)S_LEN * HID * 2;

  { int n = S_LEN * HID;
    cvt_bf16_kernel<<<(n + 255) / 256, 256, 0, stream>>>(X, Xbf, n); }
  { int n = HID * HID, n2 = HID * KV_DIM;
    cvt_transpose_kernel<<<(n + 255) / 256, 256, 0, stream>>>(Wq, Wqkv_t, HID, HID);
    cvt_transpose_kernel<<<(n2 + 255) / 256, 256, 0, stream>>>(Wk, Wqkv_t + (size_t)HID * HID, HID, KV_DIM);
    cvt_transpose_kernel<<<(n2 + 255) / 256, 256, 0, stream>>>(Wv, Wqkv_t + (size_t)(HID + KV_DIM) * HID, HID, KV_DIM);
    cvt_transpose_kernel<<<(n + 255) / 256, 256, 0, stream>>>(Wo, Wo_t, HID, HID); }

  // fused QKV projection: (2048x4096) * (4096x6144)
  gemm_bf16_kernel<<<dim3(S_LEN / 128, QKV_N / 64), 256, 0, stream>>>(
      Xbf, Wqkv_t, QKV, S_LEN, QKV_N, HID);

  { int n = S_LEN * (NH + NKV) * (HD / 2);
    rope_kernel<<<(n + 255) / 256, 256, 0, stream>>>(QKV, Qbf, Kbf);
    int nv = S_LEN * KV_DIM;
    cvt_vt_kernel<<<(nv + 255) / 256, 256, 0, stream>>>(QKV, Vt); }

  attn_pass1_kernel<<<NH * (S_LEN / 128), 256, 0, stream>>>(Qbf, Kbf, m_buf, l_buf);
  attn_pass2_kernel<<<NH * (S_LEN / 128), 256, 0, stream>>>(Qbf, Kbf, Vt, m_buf, l_buf, Obf, cspart);
  reduce_colsum_kernel<<<(NH * S_LEN + 255) / 256, 256, 0, stream>>>(cspart, cur);

  // output projection straight into d_out
  gemm_bf16_kernel<<<dim3(S_LEN / 128, HID / 64), 256, 0, stream>>>(
      Obf, Wo_t, out, S_LEN, HID, HID);

  // heavy-hitter masks into d_out tail
  topk_mask_kernel<<<NH, 256, 0, stream>>>(cur, out + (size_t)S_LEN * HID);
}
